// SpikeRNN_317827580118
// MI455X (gfx1250) — compile-verified, matrix operand order checked by reading, not run on hardware
//
#include <hip/hip_runtime.h>
#include <hip/hip_bf16.h>

// ---- problem constants (from reference) ----
#define BB   32
#define LLN  512
#define CC   256
#define TT   4
#define HIDN 1024
#define BL   (BB * LLN)      // 16384
#define MTOT (TT * BL)       // 65536

typedef __bf16 bf16;
typedef __attribute__((ext_vector_type(16))) __bf16 v16bf;
typedef __attribute__((ext_vector_type(8)))  __bf16 v8bf;
typedef __attribute__((ext_vector_type(8)))  float  v8f;
typedef __attribute__((ext_vector_type(4)))  unsigned int u32x4;
typedef __attribute__((ext_vector_type(8)))  int i32x8;
typedef __attribute__((ext_vector_type(4)))  int i32x4;

union FragAB { v16bf v; v8bf h[2]; };

// ---------------------------------------------------------------------------
// TDM: 2D tile load (tile_n rows x tile_k bf16) global -> LDS, with LDS row
// padding 64B data + 16B pad = 80B stride (pad_interval=16 DW, pad_amount=4 DW).
// Descriptor layout per CDNA5 ISA.
// ---------------------------------------------------------------------------
__device__ __forceinline__ void tdm_load_tile_2d(unsigned lds_off,
                                                 const void* gsrc,
                                                 unsigned tile_k,
                                                 unsigned tile_n,
                                                 unsigned long long stride_el) {
  unsigned long long ga = (unsigned long long)gsrc;
  u32x4 g0;
  g0.x = 1u;                                    // count=1 (valid user descriptor)
  g0.y = lds_off;                               // lds_addr[31:0]
  g0.z = (unsigned)ga;                          // global_addr low
  g0.w = (unsigned)(ga >> 32) | 0x80000000u;    // global_addr high + type=2
  i32x8 g1;
  g1[0] = (1 << 16) | (1 << 20) | (3 << 22) | (3 << 25); // 2B elems, pad 16DW->+4DW
  g1[1] = (int)(tile_k << 16);                  // tensor_dim0[15:0]
  g1[2] = (int)((tile_n & 0xFFFFu) << 16);      // tensor_dim1[15:0]
  g1[3] = (int)(tile_k << 16);                  // tile_dim0
  g1[4] = (int)(tile_n & 0xFFFFu);              // tile_dim1
  g1[5] = (int)(unsigned)(stride_el & 0xFFFFFFFFull);
  g1[6] = (int)(unsigned)(stride_el >> 32);
  g1[7] = 0;
  i32x4 zz = {0, 0, 0, 0};
#if __has_include(<hip/amd_detail/amd_gfx1250_TDM.h>)
  i32x8 z8 = {0, 0, 0, 0, 0, 0, 0, 0};
  __builtin_amdgcn_tensor_load_to_lds(g0, g1, zz, zz, z8, 0);
#else
  __builtin_amdgcn_tensor_load_to_lds(g0, g1, zz, zz, 0);
#endif
}

// ---------------------------------------------------------------------------
__global__ void zero_f32_k(float* p, int n) {
  int i = blockIdx.x * blockDim.x + threadIdx.x;
  if (i < n) p[i] = 0.0f;
}

__global__ void f32_to_bf16_k(const float* __restrict__ src, bf16* __restrict__ dst, int n) {
  int i = blockIdx.x * blockDim.x + threadIdx.x;
  if (i < n) dst[i] = (bf16)src[i];
}

// ---------------------------------------------------------------------------
// Conv(K=3 along L) + BN(eval) + LIF over T -> binary spikes bf16 [T*BL, C].
// ---------------------------------------------------------------------------
__global__ void conv_bn_lif_k(const float* __restrict__ x,
                              const float* __restrict__ cw,
                              const float* __restrict__ cb,
                              const float* __restrict__ g,
                              const float* __restrict__ be,
                              const float* __restrict__ mu,
                              const float* __restrict__ var,
                              bf16* __restrict__ spk) {
  const int bl = blockIdx.x;
  const int l  = bl & (LLN - 1);
  const int c  = threadIdx.x;
  const float* row = x + (size_t)bl * CC + c;
  const float x0 = row[0];
  const float xm = (l > 0)       ? row[-CC] : 0.0f;
  const float xp = (l < LLN - 1) ? row[ CC] : 0.0f;
  float v = 0.0f;
#pragma unroll
  for (int t = 0; t < TT; ++t) {
    float y = cw[t * 3 + 0] * xm + cw[t * 3 + 1] * x0 + cw[t * 3 + 2] * xp + cb[t];
    float inv = g[t] * rsqrtf(var[t] + 1e-5f);
    y = (y - mu[t]) * inv + be[t];
    v = 0.5f * (v + y);
    float s = (v >= 1.0f) ? 1.0f : 0.0f;
    v *= (1.0f - s);
    spk[((size_t)t * BL + bl) * CC + c] = (bf16)s;
  }
}

// ---------------------------------------------------------------------------
// TDM-fed bf16 WMMA GEMM + bias + fused LIF-over-T epilogue.
// Tile: 256 rows (4t x 64bl) x 64 cols per workgroup; 8 waves.
// Wave w owns 32 rows (t=w>>1, bl=(w&1)*32..+31) x 64 cols:
//   2 A-frags x 4 B-frags = 8 WMMAs per K-step (B frags reused across A frags).
// Waves 0-3 TDM one 64-row A t-chunk, wave 4 TDMs the B tile, into the next
// buffer while computing from the current one.
// LDS arena (bytes): A buf0 @0 (20480) | A buf1 @20480 | B buf0 @40960 (5120)
// | B buf1 @46080; the fp32 epilogue buffer (256x36x4 = 36864B) overlays @0.
// ---------------------------------------------------------------------------
__launch_bounds__(256)
__global__ void gemm_lif_k(const bf16* __restrict__ A,
                           const bf16* __restrict__ W,
                           const float* __restrict__ bias,
                           int K, int mode,
                           bf16* __restrict__ spkOut,
                           float* __restrict__ meanOut,
                           float* __restrict__ redOut) {
  __shared__ __attribute__((aligned(16))) char smem[51200];
  bf16*  const smA  = (bf16*)smem;    // staging base, bf16-element indexed
  float* const ldsO = (float*)smem;   // epilogue overlay
  const unsigned smbase = (unsigned)(unsigned long long)(void*)smem;

  const int tid   = threadIdx.x;
  const int wave  = tid >> 5;
  const int lane  = tid & 31;
  const int bl0   = blockIdx.x * 64;
  const int n0    = blockIdx.y * 64;
  const int m16   = lane & 15;
  const int khalf = lane >> 4;

  const int t_w = wave >> 1;          // t owned by this wave
  const int blw = (wave & 1) * 32;    // first bl row of this wave

  v8f acc[2][4] = {{{}, {}, {}, {}}, {{}, {}, {}, {}}};
  const int nsteps = K >> 5;

  auto issue = [&](int s, int buf) {
    const int kb = s << 5;
    if (wave < 4) {          // A chunk t=wave: 64 rows x 32 k
      const bf16* src = A + ((size_t)wave * BL + bl0) * (size_t)K + kb;
      tdm_load_tile_2d(smbase + (unsigned)buf * 20480u + (unsigned)wave * 64u * 80u,
                       src, 32u, 64u, (unsigned long long)K);
    } else if (wave == 4) {  // B tile: 64 rows x 32 k
      const bf16* src = W + (size_t)n0 * (size_t)K + kb;
      tdm_load_tile_2d(smbase + 40960u + (unsigned)buf * 5120u,
                       src, 32u, 64u, (unsigned long long)K);
    }
  };

  issue(0, 0);
  if (wave <= 4) __builtin_amdgcn_s_wait_tensorcnt(0);
  __syncthreads();

  for (int s = 0; s < nsteps; ++s) {
    const int buf = s & 1;
    if (s + 1 < nsteps) issue(s + 1, buf ^ 1);   // DMA next tile during compute

    // element offsets into the staging arena (A: buf*10240, B: 20480 + buf*2560)
    const int aBufOff = buf * 10240;
    const int bBufOff = 20480 + buf * 2560;

    FragAB af[2];
#pragma unroll
    for (int r = 0; r < 2; ++r) {
      const bf16* ap = smA + aBufOff + (t_w * 64 + blw + r * 16 + m16) * 40 + khalf * 8;
      af[r].h[0] = *(const v8bf*)ap;
      af[r].h[1] = *(const v8bf*)(ap + 16);
    }
    FragAB bfr[4];
#pragma unroll
    for (int j = 0; j < 4; ++j) {
      const bf16* p = smA + bBufOff + (j * 16 + m16) * 40 + khalf * 8;
      bfr[j].h[0] = *(const v8bf*)p;
      bfr[j].h[1] = *(const v8bf*)(p + 16);
    }
#pragma unroll
    for (int r = 0; r < 2; ++r)
#pragma unroll
      for (int j = 0; j < 4; ++j)
        acc[r][j] = __builtin_amdgcn_wmma_f32_16x16x32_bf16(
            false, af[r].v, false, bfr[j].v, (short)0, acc[r][j], false, false);

    if (s + 1 < nsteps && wave <= 4) __builtin_amdgcn_s_wait_tensorcnt(0);
    __syncthreads();
  }

  // --- epilogue in two 32-column halves (ldsO overlays the staging arena) ---
#pragma unroll
  for (int h = 0; h < 2; ++h) {
#pragma unroll
    for (int r = 0; r < 2; ++r) {
#pragma unroll
      for (int jj = 0; jj < 2; ++jj) {
        const int j    = h * 2 + jj;
        const int ncol = jj * 16 + m16;
        const float bv = bias[n0 + h * 32 + ncol];
#pragma unroll
        for (int rv = 0; rv < 8; ++rv) {
          const int mloc = wave * 32 + r * 16 + rv + (khalf << 3);
          ldsO[mloc * 36 + ncol] = acc[r][j][rv] + bv;
        }
      }
    }
    __syncthreads();

    // LIF scan over t: 64 bl x 32 n = 2048 elements / 256 threads
#pragma unroll
    for (int q = 0; q < 8; ++q) {
      const int idx = tid + q * 256;
      const int blc = idx >> 5;   // 0..63
      const int n   = idx & 31;   // 0..31
      float v = 0.0f, ssum = 0.0f;
      float sp[TT];
#pragma unroll
      for (int t = 0; t < TT; ++t) {
        const float y = ldsO[(t * 64 + blc) * 36 + n];
        v = 0.5f * (v + y);
        const float sv = (v >= 1.0f) ? 1.0f : 0.0f;
        v *= (1.0f - sv);
        sp[t] = sv;
        ssum += sv;
      }
      const int ncol = n0 + h * 32 + n;
      if (mode == 0) {
#pragma unroll
        for (int t = 0; t < TT; ++t)
          spkOut[((size_t)t * BL + bl0 + blc) * HIDN + ncol] = (bf16)sp[t];
      } else {
        const float mval = ssum * 0.25f;
        meanOut[(size_t)(bl0 + blc) * HIDN + ncol] = mval;
        const int b = (bl0 + blc) >> 9;
        atomicAdd(&redOut[b * HIDN + ncol], mval * (1.0f / (float)LLN));
      }
    }
    __syncthreads();
  }
}

// ---------------------------------------------------------------------------
extern "C" void kernel_launch(void* const* d_in, const int* in_sizes, int n_in,
                              void* d_out, int out_size, void* d_ws, size_t ws_size,
                              hipStream_t stream) {
  (void)in_sizes; (void)n_in; (void)out_size; (void)ws_size;
  const float* inputs = (const float*)d_in[0];
  const float* conv_w = (const float*)d_in[1];
  const float* conv_b = (const float*)d_in[2];
  const float* bn_g   = (const float*)d_in[3];
  const float* bn_b   = (const float*)d_in[4];
  const float* bn_m   = (const float*)d_in[5];
  const float* bn_v   = (const float*)d_in[6];
  const float* enc_w  = (const float*)d_in[7];
  const float* enc_b  = (const float*)d_in[8];
  const float* cell_w = (const float*)d_in[9];
  const float* cell_b = (const float*)d_in[10];

  // workspace: A0 32MB | A1 128MB | A2 128MB | Wenc 0.5MB | Wc 4MB
  char* ws = (char*)d_ws;
  bf16* A0   = (bf16*)ws;
  bf16* A1   = (bf16*)(ws + (size_t)MTOT * CC * 2);
  bf16* A2   = (bf16*)(ws + (size_t)MTOT * CC * 2 + (size_t)MTOT * HIDN * 2);
  bf16* Wenc = (bf16*)(ws + (size_t)MTOT * CC * 2 + 2 * (size_t)MTOT * HIDN * 2);
  bf16* Wc   = Wenc + (size_t)HIDN * CC;

  float* outMain = (float*)d_out;                       // [B,L,HID]
  float* outRed  = outMain + (size_t)BB * LLN * HIDN;   // [B,HID]

  zero_f32_k<<<(BB * HIDN + 255) / 256, 256, 0, stream>>>(outRed, BB * HIDN);
  f32_to_bf16_k<<<(HIDN * CC + 255) / 256, 256, 0, stream>>>(enc_w, Wenc, HIDN * CC);
  f32_to_bf16_k<<<(2 * HIDN * HIDN + 255) / 256, 256, 0, stream>>>(cell_w, Wc, 2 * HIDN * HIDN);
  conv_bn_lif_k<<<BL, CC, 0, stream>>>(inputs, conv_w, conv_b, bn_g, bn_b, bn_m, bn_v, A0);

  dim3 g(BL / 64, HIDN / 64);
  gemm_lif_k<<<g, 256, 0, stream>>>(A0, Wenc, enc_b, CC,   0, A1, nullptr, nullptr);
  gemm_lif_k<<<g, 256, 0, stream>>>(A1, Wc,   cell_b, HIDN, 0, A2, nullptr, nullptr);
  gemm_lif_k<<<g, 256, 0, stream>>>(A2, Wc + (size_t)HIDN * HIDN, cell_b + HIDN,
                                    HIDN, 1, nullptr, outMain, outRed);
}
